// PCNEmbeddingPredictor_85246510891344
// MI455X (gfx1250) — matrix operand order repaired and checked by reading, not compiled, not run on hardware
//
#include <hip/hip_runtime.h>
#include <math.h>

typedef __attribute__((ext_vector_type(16))) _Float16 v16h;
typedef __attribute__((ext_vector_type(8)))  float    v8f;

constexpr int kB = 64, kS = 1024, kE = 1024, kH = 2048, kNF = 8, kNH = 4, kNIT = 10;
constexpr int kR = kB * kNH * kNF;   // 2048 refine rows (B*NH*NF)

// ---------------- TDM availability ----------------
#if __has_builtin(__builtin_amdgcn_tensor_load_to_lds) && __has_builtin(__builtin_amdgcn_s_wait_tensorcnt)
  #if __has_include(<hip/amd_detail/amd_gfx1250_TDM.h>)
    #define TDM_6ARG 1
  #else
    #define TDM_5ARG 1
  #endif
#endif

#if defined(TDM_5ARG) || defined(TDM_6ARG)
typedef __attribute__((ext_vector_type(4))) unsigned int v4u_t;
typedef __attribute__((ext_vector_type(8))) int          v8i_t;
typedef __attribute__((ext_vector_type(4))) int          v4i_t;

// 2D tile load via Tensor Data Mover: gsrc points at the tile start inside an
// [Nelems x Kelems] row-major f16 tensor; tile is tileN rows x 32 cols (contig k).
// D# layout per CDNA5 ISA.
__device__ __forceinline__ void tdm_load_tile(const _Float16* gsrc, unsigned lds_addr,
                                              int Kelems, int Nelems, int tileN) {
  unsigned long long ga = (unsigned long long)(size_t)gsrc;
  v4u_t g0;
  g0[0] = 1u;                                              // count=1, user-mode flags 0
  g0[1] = lds_addr;                                        // LDS byte address
  g0[2] = (unsigned)(ga & 0xFFFFFFFFu);                    // global_addr[31:0]
  g0[3] = (unsigned)((ga >> 32) & 0x01FFFFFFu) | (2u << 30); // global_addr[56:32], type=2
  v8i_t g1;
  g1[0] = (int)(1u << 16);                                 // workgroup_mask=0, data_size=1 (2B)
  g1[1] = (int)(((unsigned)Kelems & 0xFFFFu) << 16);       // tensor_dim0[15:0]
  g1[2] = (int)((((unsigned)Kelems >> 16) & 0xFFFFu) |
                (((unsigned)Nelems & 0xFFFFu) << 16));     // tensor_dim0[31:16] | tensor_dim1[15:0]
  g1[3] = (int)((((unsigned)Nelems >> 16) & 0xFFFFu) |
                (32u << 16));                              // tensor_dim1[31:16] | tile_dim0=TK
  g1[4] = (int)((unsigned)tileN & 0xFFFFu);                // tile_dim1, tile_dim2=0
  g1[5] = (int)Kelems;                                     // tensor_dim0_stride[31:0]
  g1[6] = 0;                                               // stride[47:32], dim1_stride[15:0]
  g1[7] = 0;
  v4i_t z4 = {0, 0, 0, 0};
#if defined(TDM_6ARG)
  v8i_t z8 = {0, 0, 0, 0, 0, 0, 0, 0};
  __builtin_amdgcn_tensor_load_to_lds(g0, g1, z4, z4, z8, 0);
#else
  __builtin_amdgcn_tensor_load_to_lds(g0, g1, z4, z4, 0);
#endif
}
#endif

// ---------------- deterministic hash RNG (Box-Muller) ----------------
__device__ __forceinline__ unsigned pcg(unsigned x) {
  x = x * 747796405u + 2891336453u;
  unsigned w = ((x >> ((x >> 28) + 4u)) ^ x) * 277803737u;
  return (w >> 22) ^ w;
}
__device__ __forceinline__ float nrand(unsigned seed, unsigned idx) {
  unsigned a = pcg(seed ^ pcg(idx ^ 0x9e3779b9u));
  unsigned b = pcg(a + 0x85ebca6bu);
  float u1 = ((float)a + 1.0f) * 2.3283064365386963e-10f;   // (0,1]
  float u2 = (float)b * 2.3283064365386963e-10f;
  return sqrtf(fmaxf(-2.0f * logf(u1), 0.0f)) * cosf(6.283185307179586f * u2);
}

// ---------------- ctx = mean over S ----------------
__global__ __launch_bounds__(256) void mean_ctx_kernel(const float* __restrict__ cur,
                                                       float* __restrict__ ctx) {
  int idx = blockIdx.x * 256 + threadIdx.x;          // B*E
  if (idx >= kB * kE) return;
  int b = idx / kE, e = idx - b * kE;
  const float* p = cur + (size_t)b * kS * kE + e;
  float s = 0.f;
  for (int si = 0; si < kS; ++si) s += p[(size_t)si * kE];
  ctx[idx] = s * (1.0f / kS);
}

// ---------------- weight prep: f32 [K][N] -> f16 [N][K] (transpose + convert) ----------------
__global__ __launch_bounds__(256) void cvt_transpose_kernel(const float* __restrict__ W,
                                                            _Float16* __restrict__ WT,
                                                            int K, int N) {
  __shared__ float tile[32][33];
  const int kb = blockIdx.x * 32, nb = blockIdx.y * 32;
  const int tx = threadIdx.x & 31, ty = threadIdx.x >> 5;   // 32x8
  for (int i = ty; i < 32; i += 8)
    tile[i][tx] = W[(size_t)(kb + i) * N + nb + tx];
  __syncthreads();
  for (int i = ty; i < 32; i += 8)
    WT[(size_t)(nb + i) * K + kb + tx] = (_Float16)tile[tx][i];
}

// ---------------- WMMA GEMM: C = A[MxK](f32) * WT[NxK](f16,transposed) + bias ----------------
#define TM 128
#define TN 128
#define TK 32

union FragU { v16h v; uint4 q[2]; };

__global__ __launch_bounds__(256)
void wmma_gemm_kernel(const float* __restrict__ A, const _Float16* __restrict__ WT,
                      const float* __restrict__ bias, float* __restrict__ C,
                      int M, int N, int K, int fuse_relu) {
  __shared__ __align__(64) _Float16 sA[TM * TK];     // [m][k]  8 KB
  __shared__ __align__(64) _Float16 sB[TN * TK];     // [n][k]  8 KB

  const int bn   = blockIdx.x * TN;
  const int bm   = blockIdx.y * TM;
  const int tid  = threadIdx.x;
  const int wid  = tid >> 5;
  const int lane = tid & 31;
  const int l16  = lane & 15;
  const int half = lane >> 4;
  const int wm   = (wid & 3) * 32;    // wave row offset in tile (4 waves down)
  const int wn   = (wid >> 2) * 64;   // wave col offset in tile (2 waves across)

  v8f acc[2][4] = {};                 // 32x64 per wave -> 8 WMMAs per K-step

  // staging maps
  const int am  = tid >> 1;           // 0..127
  const int ak  = (tid & 1) << 4;     // 0 or 16
#if !(defined(TDM_5ARG) || defined(TDM_6ARG))
  const int bnr = tid >> 1;           // 0..127
  const int bks = (tid & 1) << 4;     // 0 or 16
#endif

  for (int k0 = 0; k0 < K; k0 += TK) {
    { // stage A tile, f32 -> f16 (rows past M zero-filled)
      const int gm = bm + am;
      _Float16* dst = &sA[am * TK + ak];
      if (gm < M) {
        const float* src = A + (size_t)gm * K + (k0 + ak);
        #pragma unroll
        for (int j = 0; j < 16; ++j) dst[j] = (_Float16)src[j];
      } else {
        #pragma unroll
        for (int j = 0; j < 16; ++j) dst[j] = (_Float16)0.0f;
      }
    }
#if defined(TDM_5ARG) || defined(TDM_6ARG)
    // stage B tile via Tensor Data Mover: TN x TK tile of WT, contiguous in k.
    if (tid == 0) {
      tdm_load_tile(WT + (size_t)bn * K + k0,
                    (unsigned)(size_t)(&sB[0]), K, N, TN);
      __builtin_amdgcn_s_wait_tensorcnt(0);
    }
#else
    { // manual stage: 32B contiguous copies (already f16, already transposed)
      const uint4* src = reinterpret_cast<const uint4*>(WT + (size_t)(bn + bnr) * K + k0 + bks);
      uint4* dst = reinterpret_cast<uint4*>(&sB[bnr * TK + bks]);
      dst[0] = src[0];
      dst[1] = src[1];
    }
#endif
    __syncthreads();

    // A fragments (16x32): lanes 0-15 K={0..7,16..23}, lanes 16-31 +8
    FragU a0, a1;
    {
      const _Float16* ab0 = &sA[(wm + l16) * TK];
      a0.q[0] = *reinterpret_cast<const uint4*>(ab0 + half * 8);
      a0.q[1] = *reinterpret_cast<const uint4*>(ab0 + half * 8 + 16);
      const _Float16* ab1 = &sA[(wm + 16 + l16) * TK];
      a1.q[0] = *reinterpret_cast<const uint4*>(ab1 + half * 8);
      a1.q[1] = *reinterpret_cast<const uint4*>(ab1 + half * 8 + 16);
    }
    #pragma unroll
    for (int nb = 0; nb < 4; ++nb) {
      // B fragment (32x16): lane -> col, lanes 0-15 K={0..7,16..23}, lanes 16-31 +8
      FragU bf;
      const _Float16* bb = &sB[(wn + nb * 16 + l16) * TK + half * 8];
      bf.q[0] = *reinterpret_cast<const uint4*>(bb);
      bf.q[1] = *reinterpret_cast<const uint4*>(bb + 16);
      acc[0][nb] = __builtin_amdgcn_wmma_f32_16x16x32_f16(
          false, a0.v, false, bf.v, (short)0, acc[0][nb], false, false);
      acc[1][nb] = __builtin_amdgcn_wmma_f32_16x16x32_f16(
          false, a1.v, false, bf.v, (short)0, acc[1][nb], false, false);
    }
    __syncthreads();
  }

  // C/D layout: VGPR r -> M=r (+8 for lanes 16-31), N=l16
  #pragma unroll
  for (int mi = 0; mi < 2; ++mi) {
    const int row0 = bm + wm + mi * 16 + half * 8;
    #pragma unroll
    for (int nb = 0; nb < 4; ++nb) {
      const int col = bn + wn + nb * 16 + l16;
      const float bv = bias ? bias[col] : 0.0f;
      #pragma unroll
      for (int r = 0; r < 8; ++r) {
        const int row = row0 + r;
        if (row < M) {
          float v = acc[mi][nb][r] + bv;
          if (fuse_relu) v = fmaxf(v, 0.0f);
          C[(size_t)row * N + col] = v;
        }
      }
    }
  }
}

// ---------------- row LayerNorm + ReLU (in place) ----------------
__global__ __launch_bounds__(256) void ln_relu_kernel(float* __restrict__ xm,
                                                      const float* __restrict__ g,
                                                      const float* __restrict__ be, int N) {
  const int row = blockIdx.x, tid = threadIdx.x;
  float* xr = xm + (size_t)row * N;
  __shared__ float r1[256], r2[256];
  float s = 0.f, s2 = 0.f;
  for (int i = tid; i < N; i += 256) { float v = xr[i]; s += v; s2 += v * v; }
  r1[tid] = s; r2[tid] = s2; __syncthreads();
  for (int off = 128; off > 0; off >>= 1) {
    if (tid < off) { r1[tid] += r1[tid + off]; r2[tid] += r2[tid + off]; }
    __syncthreads();
  }
  const float mean = r1[0] / N;
  const float var  = r2[0] / N - mean * mean;
  const float inv  = rsqrtf(var + 1e-5f);
  for (int i = tid; i < N; i += 256) {
    float v = (xr[i] - mean) * inv * g[i] + be[i];
    xr[i] = fmaxf(v, 0.0f);
  }
}

// ---------------- hyp = traj broadcast over NH + noise ----------------
__global__ __launch_bounds__(256) void traj_noise_kernel(const float* __restrict__ traj,
                                                         float* __restrict__ x) {
  int idx = blockIdx.x * 256 + threadIdx.x;          // kR * kE
  if (idx >= kR * kE) return;
  int r = idx / kE, e = idx - r * kE;                // r = ((b*NH)+nh)*NF + f
  int b = r / (kNH * kNF);
  int f = r % kNF;
  float v = traj[((size_t)b * kNF + f) * kE + e];
  x[idx] = v + 0.1f * nrand(42u, (unsigned)idx);
}

// ---------------- refine concat: [x | ce | neighbor] ----------------
__global__ __launch_bounds__(256) void concat_refine_kernel(const float* __restrict__ x,
                                                            const float* __restrict__ ctxh,
                                                            float* __restrict__ inp) {
  const int Wd = 3 * kE;
  int idx = blockIdx.x * 256 + threadIdx.x;
  if (idx >= kR * Wd) return;
  int r = idx / Wd, c = idx - r * Wd;
  float v;
  if (c < kE) {
    v = x[(size_t)r * kE + c];
  } else if (c < 2 * kE) {
    int b = r / (kNH * kNF);
    v = ctxh[(size_t)b * kH + (c - kE)];             // ctx_e = ctx_h[:, :E]
  } else {
    int e = c - 2 * kE;
    int f = r % kNF;
    int rp = (f == 0)       ? r : r - 1;
    int rn = (f == kNF - 1) ? r : r + 1;
    v = 0.5f * (x[(size_t)rp * kE + e] + x[(size_t)rn * kE + e]);
  }
  inp[idx] = v;
}

// ---------------- energy concat: [x | ce] ----------------
__global__ __launch_bounds__(256) void concat_energy_kernel(const float* __restrict__ x,
                                                            const float* __restrict__ ctxh,
                                                            float* __restrict__ inp) {
  const int Wd = 2 * kE;
  int idx = blockIdx.x * 256 + threadIdx.x;
  if (idx >= kR * Wd) return;
  int r = idx / Wd, c = idx - r * Wd;
  float v;
  if (c < kE) {
    v = x[(size_t)r * kE + c];
  } else {
    int b = r / (kNH * kNF);
    v = ctxh[(size_t)b * kH + (c - kE)];
  }
  inp[idx] = v;
}

// ---------------- x -= LR*u (+ small noise except last iter) ----------------
__global__ __launch_bounds__(256) void update_x_kernel(float* __restrict__ x,
                                                       const float* __restrict__ u, int it) {
  int idx = blockIdx.x * 256 + threadIdx.x;
  if (idx >= kR * kE) return;
  float v = x[idx] - 0.1f * u[idx];
  if (it < kNIT - 1) v += 0.01f * nrand(1000003u * (unsigned)(7 + it) + 13u, (unsigned)idx);
  x[idx] = v;
}

// ---------------- en_w3 dot (N=1 GEMM) ----------------
__global__ __launch_bounds__(256) void dotw3_kernel(const float* __restrict__ eh2,
                                                    const float* __restrict__ w3,
                                                    const float* __restrict__ b3,
                                                    float* __restrict__ out, int K) {
  const int row = blockIdx.x, tid = threadIdx.x;
  __shared__ float red[256];
  float s = 0.f;
  for (int i = tid; i < K; i += 256) s += eh2[(size_t)row * K + i] * w3[i];
  red[tid] = s; __syncthreads();
  for (int off = 128; off > 0; off >>= 1) {
    if (tid < off) red[tid] += red[tid + off];
    __syncthreads();
  }
  if (tid == 0) out[row] = red[0] + b3[0];
}

// ---------------- energies: coherence + 0.1*smoothness + supervision ----------------
__global__ __launch_bounds__(256) void energies_kernel(const float* __restrict__ x,
                                                       const float* __restrict__ ft,
                                                       const float* __restrict__ en3,
                                                       float* __restrict__ energ) {
  const int bh = blockIdx.x;                 // 0..B*NH-1
  const int b  = bh / kNH;
  const int tid = threadIdx.x;
  __shared__ float red[256];
  const size_t base = (size_t)bh * kNF * kE;

  float s = 0.f;                             // supervision
  for (int i = tid; i < kNF * kE; i += 256) {
    int f = i / kE, e = i - f * kE;
    float d = x[base + i] - ft[((size_t)b * kNF + f) * kE + e];
    s += d * d;
  }
  red[tid] = s; __syncthreads();
  for (int off = 128; off > 0; off >>= 1) {
    if (tid < off) red[tid] += red[tid + off];
    __syncthreads();
  }
  const float sup = red[0] / (float)(kNF * kE);
  __syncthreads();

  float smooth = 0.f;                        // smoothness (tid 0 accumulates)
  for (int j = 0; j < kNF - 1; ++j) {
    float sj = 0.f;
    for (int e = tid; e < kE; e += 256) {
      float d = x[base + (size_t)(j + 1) * kE + e] - x[base + (size_t)j * kE + e];
      sj += d * d;
    }
    red[tid] = sj; __syncthreads();
    for (int off = 128; off > 0; off >>= 1) {
      if (tid < off) red[tid] += red[tid + off];
      __syncthreads();
    }
    if (tid == 0) smooth += sqrtf(red[0]);
    __syncthreads();
  }

  if (tid == 0) {
    float coh = 0.f;
    for (int f = 0; f < kNF; ++f) coh += en3[bh * kNF + f];
    coh /= (float)kNF;
    energ[bh] = coh + 0.1f * (smooth / (float)(kNF - 1)) + sup;
  }
}

// ---------------- softmax over NH + weighted sum ----------------
__global__ __launch_bounds__(256) void combine_kernel(const float* __restrict__ x,
                                                      const float* __restrict__ energ,
                                                      float* __restrict__ out) {
  int idx = blockIdx.x * 256 + threadIdx.x;          // B*NF*E
  if (idx >= kB * kNF * kE) return;
  int b  = idx / (kNF * kE);
  int fe = idx - b * (kNF * kE);
  float e0 = energ[b * kNH + 0], e1 = energ[b * kNH + 1];
  float e2 = energ[b * kNH + 2], e3 = energ[b * kNH + 3];
  float m = fminf(fminf(e0, e1), fminf(e2, e3));
  float w0 = expf(-(e0 - m)), w1 = expf(-(e1 - m));
  float w2 = expf(-(e2 - m)), w3 = expf(-(e3 - m));
  float inv = 1.0f / (w0 + w1 + w2 + w3);
  float acc = 0.f;
  acc += x[((size_t)(b * kNH + 0) * kNF * kE) + fe] * w0;
  acc += x[((size_t)(b * kNH + 1) * kNF * kE) + fe] * w1;
  acc += x[((size_t)(b * kNH + 2) * kNF * kE) + fe] * w2;
  acc += x[((size_t)(b * kNH + 3) * kNF * kE) + fe] * w3;
  out[idx] = acc * inv;
}

// ---------------- host ----------------
static inline void gemm(const float* A, const _Float16* WT, const float* bias, float* C,
                        int M, int N, int K, int relu, hipStream_t s) {
  dim3 grid(N / TN, (M + TM - 1) / TM);
  wmma_gemm_kernel<<<grid, 256, 0, s>>>(A, WT, bias, C, M, N, K, relu);
}
static inline void prep_w(const float* W, _Float16* WT, int K, int N, hipStream_t s) {
  dim3 grid(K / 32, N / 32);
  cvt_transpose_kernel<<<grid, 256, 0, s>>>(W, WT, K, N);
}

extern "C" void kernel_launch(void* const* d_in, const int* in_sizes, int n_in,
                              void* d_out, int out_size, void* d_ws, size_t ws_size,
                              hipStream_t stream) {
  (void)in_sizes; (void)n_in; (void)out_size; (void)ws_size;
  const float* cur   = (const float*)d_in[0];
  const float* ftrue = (const float*)d_in[1];
  const float* ce_w1 = (const float*)d_in[2];
  const float* ce_b1 = (const float*)d_in[3];
  const float* ce_g  = (const float*)d_in[4];
  const float* ce_be = (const float*)d_in[5];
  const float* ce_w2 = (const float*)d_in[6];
  const float* ce_b2 = (const float*)d_in[7];
  const float* tp_w1 = (const float*)d_in[8];
  const float* tp_b1 = (const float*)d_in[9];
  const float* tp_g  = (const float*)d_in[10];
  const float* tp_be = (const float*)d_in[11];
  const float* tp_w2 = (const float*)d_in[12];
  const float* tp_b2 = (const float*)d_in[13];
  const float* en_w1 = (const float*)d_in[14];
  const float* en_b1 = (const float*)d_in[15];
  const float* en_g  = (const float*)d_in[16];
  const float* en_be = (const float*)d_in[17];
  const float* en_w2 = (const float*)d_in[18];
  const float* en_b2 = (const float*)d_in[19];
  const float* en_w3 = (const float*)d_in[20];
  const float* en_b3 = (const float*)d_in[21];
  const float* rn_w1 = (const float*)d_in[22];
  const float* rn_b1 = (const float*)d_in[23];
  const float* rn_g  = (const float*)d_in[24];
  const float* rn_be = (const float*)d_in[25];
  const float* rn_w2 = (const float*)d_in[26];
  const float* rn_b2 = (const float*)d_in[27];

  char* wsb = (char*)d_ws;
  size_t off = 0;
  auto alloc = [&](size_t bytes) -> char* {
    char* p = wsb + off;
    off += (bytes + 255) & ~(size_t)255;
    return p;
  };
  float* ctx   = (float*)alloc((size_t)kB * kE * 4);
  float* h     = (float*)alloc((size_t)kB * kH * 4);
  float* ctxh  = (float*)alloc((size_t)kB * kH * 4);
  float* t     = (float*)alloc((size_t)kB * kH * 4);
  float* traj  = (float*)alloc((size_t)kB * kNF * kE * 4);
  float* x     = (float*)alloc((size_t)kR * kE * 4);
  float* inp   = (float*)alloc((size_t)kR * 3 * kE * 4);   // reused for energy concat
  float* uh    = (float*)alloc((size_t)kR * kH * 4);       // reused as eh1
  float* u     = (float*)alloc((size_t)kR * kE * 4);       // reused as eh2
  float* en3   = (float*)alloc((size_t)kR * 4);
  float* energ = (float*)alloc((size_t)kB * kNH * 4);
  // f16 transposed weights [N][K]
  _Float16* wt_ce1 = (_Float16*)alloc((size_t)kE * kH * 2);
  _Float16* wt_ce2 = (_Float16*)alloc((size_t)kH * kH * 2);
  _Float16* wt_tp1 = (_Float16*)alloc((size_t)kH * kH * 2);
  _Float16* wt_tp2 = (_Float16*)alloc((size_t)kH * kNF * kE * 2);
  _Float16* wt_en1 = (_Float16*)alloc((size_t)2 * kE * kH * 2);
  _Float16* wt_en2 = (_Float16*)alloc((size_t)kH * (kH / 2) * 2);
  _Float16* wt_rn1 = (_Float16*)alloc((size_t)3 * kE * kH * 2);
  _Float16* wt_rn2 = (_Float16*)alloc((size_t)kH * kE * 2);

  // 0) weight prep: convert+transpose once per launch (stays resident in L2)
  prep_w(ce_w1, wt_ce1, kE,     kH,        stream);
  prep_w(ce_w2, wt_ce2, kH,     kH,        stream);
  prep_w(tp_w1, wt_tp1, kH,     kH,        stream);
  prep_w(tp_w2, wt_tp2, kH,     kNF * kE,  stream);
  prep_w(en_w1, wt_en1, 2 * kE, kH,        stream);
  prep_w(en_w2, wt_en2, kH,     kH / 2,    stream);
  prep_w(rn_w1, wt_rn1, 3 * kE, kH,        stream);
  prep_w(rn_w2, wt_rn2, kH,     kE,        stream);

  // 1) ctx = mean over S (HBM-bound: 268 MB read ~ 12 us at 23.3 TB/s)
  mean_ctx_kernel<<<(kB * kE) / 256, 256, 0, stream>>>(cur, ctx);

  // 2) ce head
  gemm(ctx, wt_ce1, ce_b1, h, kB, kH, kE, 0, stream);
  ln_relu_kernel<<<kB, 256, 0, stream>>>(h, ce_g, ce_be, kH);
  gemm(h, wt_ce2, ce_b2, ctxh, kB, kH, kH, 0, stream);

  // 3) tp head -> trajectory
  gemm(ctxh, wt_tp1, tp_b1, t, kB, kH, kH, 0, stream);
  ln_relu_kernel<<<kB, 256, 0, stream>>>(t, tp_g, tp_be, kH);
  gemm(t, wt_tp2, tp_b2, traj, kB, kNF * kE, kH, 0, stream);

  // 4) hypotheses = traj + noise
  traj_noise_kernel<<<(kR * kE) / 256, 256, 0, stream>>>(traj, x);

  // 5) refine loop (dominant compute: ~344 GFLOP via WMMA)
  for (int it = 0; it < kNIT; ++it) {
    concat_refine_kernel<<<(kR * 3 * kE) / 256, 256, 0, stream>>>(x, ctxh, inp);
    gemm(inp, wt_rn1, rn_b1, uh, kR, kH, 3 * kE, 0, stream);
    ln_relu_kernel<<<kR, 256, 0, stream>>>(uh, rn_g, rn_be, kH);
    gemm(uh, wt_rn2, rn_b2, u, kR, kE, kH, 0, stream);
    update_x_kernel<<<(kR * kE) / 256, 256, 0, stream>>>(x, u, it);
  }

  // 6) energy head
  concat_energy_kernel<<<(kR * 2 * kE) / 256, 256, 0, stream>>>(x, ctxh, inp);
  gemm(inp, wt_en1, en_b1, uh, kR, kH, 2 * kE, 0, stream);
  ln_relu_kernel<<<kR, 256, 0, stream>>>(uh, en_g, en_be, kH);
  gemm(uh, wt_en2, en_b2, u, kR, kH / 2, kH, 1, stream);           // fused ReLU
  dotw3_kernel<<<kR, 256, 0, stream>>>(u, en_w3, en_b3, en3, kH / 2);
  energies_kernel<<<kB * kNH, 256, 0, stream>>>(x, ftrue, en3, energ);

  // 7) softmax over hypotheses + weighted sum -> predictions
  combine_kernel<<<(kB * kNF * kE) / 256, 256, 0, stream>>>(x, energ, (float*)d_out);
}
